// MonolithicDecoderLayer_66623532695921
// MI455X (gfx1250) — matrix operand order repaired and checked by reading, not compiled, not run on hardware
//
#include <hip/hip_runtime.h>
#include <stdint.h>

// ---------------- problem constants ----------------
#define T_    2048
#define D_    2048
#define H_    16
#define DN_   128
#define DR_   64
#define DV_   128
#define QL_   768
#define KVL_  512
#define HI_   16
#define DI_   128
#define TOPK_ 256
#define E_    8
#define MI_   1024
#define EPS_  1e-6f
#define RSF_  2.5f
#define NEG_  (-1e30f)
#define KVW_  (KVL_ + DR_)          // 576
#define QKVN_ (QL_ + KVL_ + DR_)    // 1344
#define QBN_  (H_ * (DN_ + DR_))    // 3072
#define SCALE_  0.07216878364870322f   // (DN+DR)^-0.5
#define ISCALE_ 0.08838834764831845f   // DI^-0.5

// ---------------- feature probes (compile-safe) ----------------
#if defined(__has_builtin)
#  if __has_builtin(__builtin_amdgcn_global_load_async_to_lds_b128) && \
      __has_builtin(__builtin_amdgcn_s_wait_asynccnt)
#    define USE_ASYNC_LDS 1
#  else
#    define USE_ASYNC_LDS 0
#  endif
#else
#  define USE_ASYNC_LDS 0
#endif

// ---------------- WMMA types ----------------
typedef __attribute__((ext_vector_type(16))) __bf16 v16bf;
typedef __attribute__((ext_vector_type(8)))  float  v8f;
typedef __attribute__((ext_vector_type(4)))  int    v4i_;
#define AS1_ __attribute__((address_space(1)))
#define AS3_ __attribute__((address_space(3)))

union Frag16 { unsigned int u[8]; v16bf v; };

__device__ __forceinline__ unsigned short f2bf(float f) {
  unsigned u = __float_as_uint(f);
  unsigned r = u + 0x7FFFu + ((u >> 16) & 1u);   // RNE
  return (unsigned short)(r >> 16);
}

// A fragment: 16x32 bf16, rows m0..m0+15 of LDS tile (row-major, pitch 32).
// ISA layout: lanes 0-15 hold K 0..7 / 16..23, lanes 16-31 hold K 8..15 / 24..31.
__device__ __forceinline__ v16bf load_a_frag(const unsigned short* As, int m0, int lane) {
  int r = m0 + (lane & 15);
  int koff = (lane >> 4) * 8;
  const uint4 a0 = *(const uint4*)(As + r * 32 + koff);
  const uint4 a1 = *(const uint4*)(As + r * 32 + koff + 16);
  Frag16 f;
  f.u[0] = a0.x; f.u[1] = a0.y; f.u[2] = a0.z; f.u[3] = a0.w;
  f.u[4] = a1.x; f.u[5] = a1.y; f.u[6] = a1.z; f.u[7] = a1.w;
  return f.v;
}
// B fragment: 32x16 bf16 from LDS tile stored N-major (Bs[n][k], pitch 32):
// lanes 0-15 -> K 0..7 / 16..23, lanes 16-31 -> K 8..15 / 24..31, col = lane&15.
__device__ __forceinline__ v16bf load_b_frag(const unsigned short* Bs, int n0, int lane) {
  int n = n0 + (lane & 15);
  int kb = (lane >> 4) * 8;
  const uint4 b0 = *(const uint4*)(Bs + n * 32 + kb);
  const uint4 b1 = *(const uint4*)(Bs + n * 32 + kb + 16);
  Frag16 f;
  f.u[0] = b0.x; f.u[1] = b0.y; f.u[2] = b0.z; f.u[3] = b0.w;
  f.u[4] = b1.x; f.u[5] = b1.y; f.u[6] = b1.z; f.u[7] = b1.w;
  return f.v;
}

// ---------------- generic WMMA GEMM (all-bf16 operands) ----------------
// C[M,N] = epilogue( Ab[M,K](bf16 row-major) * Bt[N,K](bf16 N-major) )
// 128x128 block, 8 waves * (2x4) 16x16 tiles -> 8 wmma / K-step.
// Double-buffered async global->LDS staging (raw bf16 copies, no conversion).
// Requires M%128==0, K%32==0; ragged N handled by guards.
__global__ __launch_bounds__(256) void k_gemm(
    const unsigned short* __restrict__ Ab, const unsigned short* __restrict__ Bt,
    float* __restrict__ C, unsigned short* __restrict__ Cb,
    int M, int N, int K, long long lda, long long ldb, long long ldc,
    float alpha, int beta,
    const float* __restrict__ rowScale, int rsStride, int act,
    long long sA, long long sB, long long sC) {
  __shared__ __align__(16) unsigned short As[2][128 * 32];   // 2 x 8 KB
  __shared__ __align__(16) unsigned short Bs[2][128 * 32];   // 2 x 8 KB, n-major
  Ab += (long long)blockIdx.z * sA;
  Bt += (long long)blockIdx.z * sB;
  C  += (long long)blockIdx.z * sC;
  if (Cb) Cb += (long long)blockIdx.z * sC;
  const int tid = threadIdx.x, lane = tid & 31, wave = tid >> 5;
  const int mB = blockIdx.y * 128, nB = blockIdx.x * 128;
  const int mrow = (wave >> 1) * 32;   // 0,32,64,96
  const int ncol = (wave & 1) * 64;    // 0,64
  v8f acc[2][4] = {};

  const int rn = tid >> 1;           // 0..127: tile row for both A and B staging
  const int rc = (tid & 1) * 16;     // 0 / 16 (16 bf16 = two 16B chunks)

  auto stage = [&](int p, int kt) {
    // ---- A row (always in range: M % 128 == 0) ----
    const unsigned short* ap = Ab + (long long)(mB + rn) * lda + kt + rc;
    unsigned short* ad = &As[p][rn * 32 + rc];
    // ---- B row (guarded) ----
    int n = nB + rn;
    unsigned short* bd = &Bs[p][rn * 32 + rc];
#if USE_ASYNC_LDS
    __builtin_amdgcn_global_load_async_to_lds_b128((AS1_ v4i_*)ap,       (AS3_ v4i_*)ad,       0, 0);
    __builtin_amdgcn_global_load_async_to_lds_b128((AS1_ v4i_*)(ap + 8), (AS3_ v4i_*)(ad + 8), 0, 0);
    if (n < N) {
      const unsigned short* bp = Bt + (long long)n * ldb + kt + rc;
      __builtin_amdgcn_global_load_async_to_lds_b128((AS1_ v4i_*)bp,       (AS3_ v4i_*)bd,       0, 0);
      __builtin_amdgcn_global_load_async_to_lds_b128((AS1_ v4i_*)(bp + 8), (AS3_ v4i_*)(bd + 8), 0, 0);
    } else {
      uint4 z; z.x = z.y = z.z = z.w = 0u;
      *(uint4*)bd = z; *(uint4*)(bd + 8) = z;
    }
#else
    *(uint4*)ad       = *(const uint4*)ap;
    *(uint4*)(ad + 8) = *(const uint4*)(ap + 8);
    if (n < N) {
      const unsigned short* bp = Bt + (long long)n * ldb + kt + rc;
      *(uint4*)bd       = *(const uint4*)bp;
      *(uint4*)(bd + 8) = *(const uint4*)(bp + 8);
    } else {
      uint4 z; z.x = z.y = z.z = z.w = 0u;
      *(uint4*)bd = z; *(uint4*)(bd + 8) = z;
    }
#endif
  };

  stage(0, 0);
  int p = 0;
  for (int kt = 0; kt < K; kt += 32) {
    const bool more = (kt + 32) < K;
    if (more) stage(p ^ 1, kt + 32);         // prefetch next tile into other buffer
#if USE_ASYNC_LDS
    // async loads complete in order: waiting <=4 guarantees the current tile's
    // 4 per-wave loads landed while the next tile's 4 remain in flight.
    if (more) __builtin_amdgcn_s_wait_asynccnt(4);
    else      __builtin_amdgcn_s_wait_asynccnt(0);
#endif
    __syncthreads();
    const unsigned short* Asp = As[p];
    const unsigned short* Bsp = Bs[p];
    v16bf a0 = load_a_frag(Asp, mrow, lane);
    v16bf a1 = load_a_frag(Asp, mrow + 16, lane);
    v16bf b0 = load_b_frag(Bsp, ncol, lane);
    v16bf b1 = load_b_frag(Bsp, ncol + 16, lane);
    v16bf b2 = load_b_frag(Bsp, ncol + 32, lane);
    v16bf b3 = load_b_frag(Bsp, ncol + 48, lane);
    acc[0][0] = __builtin_amdgcn_wmma_f32_16x16x32_bf16(false, a0, false, b0, (short)0, acc[0][0], false, false);
    acc[0][1] = __builtin_amdgcn_wmma_f32_16x16x32_bf16(false, a0, false, b1, (short)0, acc[0][1], false, false);
    acc[0][2] = __builtin_amdgcn_wmma_f32_16x16x32_bf16(false, a0, false, b2, (short)0, acc[0][2], false, false);
    acc[0][3] = __builtin_amdgcn_wmma_f32_16x16x32_bf16(false, a0, false, b3, (short)0, acc[0][3], false, false);
    acc[1][0] = __builtin_amdgcn_wmma_f32_16x16x32_bf16(false, a1, false, b0, (short)0, acc[1][0], false, false);
    acc[1][1] = __builtin_amdgcn_wmma_f32_16x16x32_bf16(false, a1, false, b1, (short)0, acc[1][1], false, false);
    acc[1][2] = __builtin_amdgcn_wmma_f32_16x16x32_bf16(false, a1, false, b2, (short)0, acc[1][2], false, false);
    acc[1][3] = __builtin_amdgcn_wmma_f32_16x16x32_bf16(false, a1, false, b3, (short)0, acc[1][3], false, false);
    __syncthreads();
    p ^= 1;
  }
  // epilogue (C layout: VGPR v -> M = m0 + v + (lane>=16 ? 8:0), N = lane&15)
  const int hi8 = (lane >> 4) ? 8 : 0;
  #pragma unroll
  for (int im = 0; im < 2; ++im) {
    #pragma unroll
    for (int jn = 0; jn < 4; ++jn) {
      int col = nB + ncol + jn * 16 + (lane & 15);
      if (col >= N) continue;
      int rbase = mB + mrow + im * 16 + hi8;
      #pragma unroll
      for (int v = 0; v < 8; ++v) {
        int row = rbase + v;
        float val = acc[im][jn][v];
        if (act) val = fmaxf(val, 0.f);
        val *= alpha;
        if (rowScale) val *= rowScale[(long long)row * rsStride];
        float* cp = C + (long long)row * ldc + col;
        float v2 = beta ? (*cp + val) : val;
        *cp = v2;
        if (Cb) Cb[(long long)row * ldc + col] = f2bf(v2);   // bf16 shadow for chained GEMMs
      }
    }
  }
}

// ---------------- convert / transpose to bf16 ----------------
// out[c*R + r] = bf16(in[r*ldIn + c]) : (R x C) fp32 -> (C x R) bf16
__global__ __launch_bounds__(256) void k_convT(
    const float* __restrict__ in, unsigned short* __restrict__ out,
    int R, int C, long long ldIn, long long sIn, long long sOut) {
  __shared__ unsigned short tile[32][33];
  in  += (long long)blockIdx.z * sIn;
  out += (long long)blockIdx.z * sOut;
  int r0 = blockIdx.y * 32, c0 = blockIdx.x * 32;
  int tx = threadIdx.x & 31, ty = threadIdx.x >> 5;
  for (int i = ty; i < 32; i += 8) {
    int r = r0 + i, c = c0 + tx;
    float v = (r < R && c < C) ? in[(long long)r * ldIn + c] : 0.f;
    tile[i][tx] = f2bf(v);
  }
  __syncthreads();
  for (int i = ty; i < 32; i += 8) {
    int c = c0 + i, r = r0 + tx;
    if (c < C && r < R) out[(long long)c * R + r] = tile[tx][i];
  }
}

// ---------------- reductions ----------------
__device__ __forceinline__ float block_sum(float v, float* shm) {
  #pragma unroll
  for (int o = 16; o; o >>= 1) v += __shfl_xor(v, o);
  int lane = threadIdx.x & 31, wave = threadIdx.x >> 5;
  int nw = blockDim.x >> 5;
  if (lane == 0) shm[wave] = v;
  __syncthreads();
  float tot = 0.f;
  for (int w = 0; w < nw; ++w) tot += shm[w];
  __syncthreads();
  return tot;
}
__device__ __forceinline__ float block_max(float v, float* shm) {
  #pragma unroll
  for (int o = 16; o; o >>= 1) v = fmaxf(v, __shfl_xor(v, o));
  int lane = threadIdx.x & 31, wave = threadIdx.x >> 5;
  int nw = blockDim.x >> 5;
  if (lane == 0) shm[wave] = v;
  __syncthreads();
  float m = shm[0];
  for (int w = 1; w < nw; ++w) m = fmaxf(m, shm[w]);
  __syncthreads();
  return m;
}

// ---------------- norms (optional bf16 shadow output) ----------------
__global__ __launch_bounds__(256) void k_rmsnorm(
    const float* __restrict__ x, const float* __restrict__ w,
    float* __restrict__ y, unsigned short* __restrict__ yb,
    int d, long long xp, long long yp, long long ypb) {
  __shared__ float shm[8];
  const float* xr = x + (long long)blockIdx.x * xp;
  float* yr = y + (long long)blockIdx.x * yp;
  unsigned short* ybr = yb ? yb + (long long)blockIdx.x * ypb : nullptr;
  float ss = 0.f;
  for (int i = threadIdx.x; i < d; i += blockDim.x) { float v = xr[i]; ss += v * v; }
  float tot = block_sum(ss, shm);
  float sc = rsqrtf(tot / d + EPS_);
  for (int i = threadIdx.x; i < d; i += blockDim.x) {
    float o = xr[i] * sc * w[i];
    yr[i] = o;
    if (ybr) ybr[i] = f2bf(o);
  }
}

__global__ __launch_bounds__(256) void k_layernorm(
    float* __restrict__ x, const float* __restrict__ w, const float* __restrict__ b,
    int d, long long p) {
  __shared__ float shm[8];
  float* xr = x + (long long)blockIdx.x * p;
  float s = 0.f, s2 = 0.f;
  for (int i = threadIdx.x; i < d; i += blockDim.x) { float v = xr[i]; s += v; s2 += v * v; }
  float S  = block_sum(s, shm);
  float S2 = block_sum(s2, shm);
  float mu = S / d;
  float rs = rsqrtf(S2 / d - mu * mu + EPS_);
  for (int i = threadIdx.x; i < d; i += blockDim.x) xr[i] = (xr[i] - mu) * rs * w[i] + b[i];
}

// ---------------- RoPE (half-split; optional bf16 shadow) ----------------
__global__ void k_rope(const float* __restrict__ in, float* __restrict__ out,
                       unsigned short* __restrict__ outb, const int* __restrict__ pos,
                       long long inRow, long long inHead, long long outRow, long long outHead,
                       long long obRow, long long obHead, int d, float scale) {
  int t = blockIdx.x, hh = blockIdx.y, i = threadIdx.x;
  int half = d >> 1;
  if (i >= half) return;
  const float* xi = in + (long long)t * inRow + (long long)hh * inHead;
  float* xo = out + (long long)t * outRow + (long long)hh * outHead;
  float inv = __powf(10000.f, -(2.f * i) / (float)d);
  float f = (float)pos[t] * inv, sn, c;
  __sincosf(f, &sn, &c);
  float x1 = xi[i], x2 = xi[i + half];
  float y1 = (x1 * c - x2 * sn) * scale;
  float y2 = (x2 * c + x1 * sn) * scale;
  xo[i] = y1; xo[i + half] = y2;
  if (outb) {
    unsigned short* xb = outb + (long long)t * obRow + (long long)hh * obHead;
    xb[i] = f2bf(y1); xb[i + half] = f2bf(y2);
  }
}

// ---------------- top-k threshold -> bitmask ----------------
__global__ __launch_bounds__(256) void k_topk_mask(
    const float* __restrict__ il, const int* __restrict__ pos, unsigned* __restrict__ mask) {
  __shared__ unsigned keys[T_];
  __shared__ int redc[8];
  int t = blockIdx.x;
  int pt = pos[t];
  for (int s = threadIdx.x; s < T_; s += 256) {
    float v = (pos[s] <= pt) ? il[(long long)t * T_ + s] : NEG_;
    unsigned u = __float_as_uint(v);
    u = (u & 0x80000000u) ? ~u : (u | 0x80000000u);   // order-preserving key
    keys[s] = u;
  }
  __syncthreads();
  unsigned prefix = 0u;
  for (int b = 31; b >= 0; --b) {
    unsigned cand = prefix | (1u << b);
    int cnt = 0;
    for (int s = threadIdx.x; s < T_; s += 256) cnt += (keys[s] >= cand) ? 1 : 0;
    #pragma unroll
    for (int o = 16; o; o >>= 1) cnt += __shfl_xor(cnt, o);
    if ((threadIdx.x & 31) == 0) redc[threadIdx.x >> 5] = cnt;
    __syncthreads();
    if (threadIdx.x == 0) { int tot = 0; for (int w = 0; w < 8; ++w) tot += redc[w]; redc[0] = tot; }
    __syncthreads();
    if (redc[0] >= TOPK_) prefix = cand;
    __syncthreads();
  }
  for (int w = threadIdx.x; w < T_ / 32; w += 256) {
    unsigned bits = 0u;
    for (int j = 0; j < 32; ++j) {
      int s = w * 32 + j;
      if ((pos[s] <= pt) && (keys[s] >= prefix)) bits |= (1u << j);
    }
    mask[(long long)t * (T_ / 32) + w] = bits;
  }
}

// ---------------- masked softmax (writes fp32 + bf16 probs) ----------------
__global__ __launch_bounds__(256) void k_softmax_mask(
    float* __restrict__ sc, unsigned short* __restrict__ ob,
    const unsigned* __restrict__ mask) {
  __shared__ float buf[T_];
  __shared__ float shm[8];
  long long t = blockIdx.x;
  float* row = sc + t * T_;
  unsigned short* brow = ob + t * T_;
  const unsigned* mrow = mask + t * (T_ / 32);
  float mx = -3.0e38f;
  for (int s = threadIdx.x; s < T_; s += 256) {
    bool ok = (mrow[s >> 5] >> (s & 31)) & 1u;
    float v = ok ? row[s] : NEG_;
    buf[s] = v;
    mx = fmaxf(mx, v);
  }
  __syncthreads();
  float M = block_max(mx, shm);
  float sum = 0.f;
  for (int s = threadIdx.x; s < T_; s += 256) {
    float v = buf[s];
    float e = (v > 0.5f * NEG_) ? __expf(v - M) : 0.f;
    buf[s] = e; sum += e;
  }
  float S = block_sum(sum, shm);
  float inv = 1.f / S;
  for (int s = threadIdx.x; s < T_; s += 256) {
    float pv = buf[s] * inv;
    row[s] = pv;
    brow[s] = f2bf(pv);
  }
}

// ---------------- gating: softmax + top-2 -> per-expert weights ----------------
__global__ __launch_bounds__(256) void k_gate(
    const float* __restrict__ h, const float* __restrict__ Wg, float* __restrict__ we) {
  __shared__ float shm[8];
  __shared__ float logits[E_];
  long long t = blockIdx.x;
  float a[E_] = {};
  for (int i = threadIdx.x; i < D_; i += 256) {
    float x = h[t * D_ + i];
    const float* wr = Wg + (long long)i * E_;
    #pragma unroll
    for (int e = 0; e < E_; ++e) a[e] += x * wr[e];
  }
  for (int e = 0; e < E_; ++e) {
    float s = block_sum(a[e], shm);
    if (threadIdx.x == 0) logits[e] = s;
  }
  __syncthreads();
  if (threadIdx.x == 0) {
    float mx = logits[0];
    for (int e = 1; e < E_; ++e) mx = fmaxf(mx, logits[e]);
    float p[E_], s = 0.f;
    for (int e = 0; e < E_; ++e) { p[e] = __expf(logits[e] - mx); s += p[e]; }
    for (int e = 0; e < E_; ++e) p[e] /= s;
    int i1 = 0; for (int e = 1; e < E_; ++e) if (p[e] > p[i1]) i1 = e;
    int i2 = (i1 == 0) ? 1 : 0;
    for (int e = 0; e < E_; ++e) if (e != i1 && p[e] > p[i2]) i2 = e;
    float ts = p[i1] + p[i2];
    for (int e = 0; e < E_; ++e)
      we[t * E_ + e] = (e == i1) ? p[i1] / ts : (e == i2) ? p[i2] / ts : 0.f;
  }
}

// ---------------- elementwise ----------------
__global__ void k_silu_mul(float* __restrict__ a, const float* __restrict__ b,
                           unsigned short* __restrict__ ab, long long n) {
  long long i = (long long)blockIdx.x * blockDim.x + threadIdx.x;
  if (i < n) {
    float x = a[i];
    float r = (x / (1.f + __expf(-x))) * b[i];
    a[i] = r;
    ab[i] = f2bf(r);
  }
}
__global__ void k_add(const float* __restrict__ a, const float* __restrict__ b,
                      float* __restrict__ c, long long n) {
  long long i = (long long)blockIdx.x * blockDim.x + threadIdx.x;
  if (i < n) c[i] = a[i] + b[i];
}
__global__ void k_combine(const float* __restrict__ r, const float* __restrict__ s,
                          float* __restrict__ o, long long n) {
  long long i = (long long)blockIdx.x * blockDim.x + threadIdx.x;
  if (i < n) o[i] = r[i] * RSF_ + s[i];
}
__global__ void k_copy_qpe(const float* __restrict__ q, float* __restrict__ qf,
                           unsigned short* __restrict__ qfb) {
  long long t = blockIdx.x; int h = blockIdx.y; int i = threadIdx.x; // 64 threads
  long long dsti = t * (long long)(H_ * KVW_) + h * KVW_ + KVL_ + i;
  float v = q[t * (long long)QBN_ + h * (DN_ + DR_) + DN_ + i];
  qf[dsti] = v;
  qfb[dsti] = f2bf(v);
}

// ---------------- host side ----------------
static inline void gemm(hipStream_t st, const unsigned short* Ab, const unsigned short* Bt,
    float* C, unsigned short* Cb,
    int M, int N, int K, long long lda, long long ldb, long long ldc,
    float alpha = 1.f, int beta = 0,
    const float* rowScale = nullptr, int rsStride = 1, int act = 0,
    int batch = 1, long long sA = 0, long long sB = 0, long long sC = 0) {
  dim3 g((N + 127) / 128, M / 128, batch), b(256);
  k_gemm<<<g, b, 0, st>>>(Ab, Bt, C, Cb, M, N, K, lda, ldb, ldc, alpha, beta,
                          rowScale, rsStride, act, sA, sB, sC);
}

extern "C" void kernel_launch(void* const* d_in, const int* in_sizes, int n_in,
                              void* d_out, int out_size, void* d_ws, size_t ws_size,
                              hipStream_t stream) {
  (void)in_sizes; (void)n_in; (void)out_size; (void)ws_size;
  const int*   pos     = (const int*)d_in[0];
  const float* hidden  = (const float*)d_in[1];
  const float* w_in    = (const float*)d_in[2];
  const float* w_post  = (const float*)d_in[3];
  const float* W_qkv_a = (const float*)d_in[4];
  const float* w_qa    = (const float*)d_in[5];
  const float* w_kva   = (const float*)d_in[6];
  const float* W_qb    = (const float*)d_in[7];
  const float* W_ik    = (const float*)d_in[8];
  const float* ik_ln_w = (const float*)d_in[9];
  const float* ik_ln_b = (const float*)d_in[10];
  const float* W_iw    = (const float*)d_in[11];
  const float* W_iqb   = (const float*)d_in[12];
  const float* W_UKT   = (const float*)d_in[13];
  const float* W_UV    = (const float*)d_in[14];
  const float* W_o     = (const float*)d_in[15];
  const float* W_gate  = (const float*)d_in[16];
  const float* We1     = (const float*)d_in[17];
  const float* We3     = (const float*)d_in[18];
  const float* We2     = (const float*)d_in[19];
  const float* Ws1     = (const float*)d_in[20];
  const float* Ws3     = (const float*)d_in[21];
  const float* Ws2     = (const float*)d_in[22];

  float* out   = (float*)d_out;                 // (T, D)
  float* resid = out + (long long)T_ * D_;      // second tuple element (T, D)

  // workspace bump allocator (16B-aligned)
  float* base = (float*)d_ws;
  size_t off = 0;
  auto alloc  = [&](size_t n) { float* p = base + off; off += (n + 3) & ~size_t(3); return p; };
  auto allocU = [&](size_t n) { return (unsigned short*)alloc((n + 1) / 2); };
  // fp32 activations
  float* h_   = alloc((size_t)T_ * D_);
  float* qkv  = alloc((size_t)T_ * QKVN_);
  float* qcn  = alloc((size_t)T_ * QL_);
  float* kv   = alloc((size_t)T_ * KVW_);
  float* ik   = alloc((size_t)T_ * DI_);
  float* iw   = alloc((size_t)T_ * HI_);
  float* q    = alloc((size_t)T_ * QBN_);
  float* iq   = alloc((size_t)T_ * HI_ * DI_);
  float* il   = alloc((size_t)T_ * T_);              // reused as per-head score buffer
  unsigned* mbits = (unsigned*)alloc((size_t)T_ * (T_ / 32));
  float* qf   = alloc((size_t)T_ * H_ * KVW_);
  float* ctx  = alloc((size_t)T_ * H_ * KVL_);
  float* ov   = alloc((size_t)T_ * H_ * DV_);
  float* oproj= alloc((size_t)T_ * D_);
  float* h2   = alloc((size_t)T_ * D_);
  float* we   = alloc((size_t)T_ * E_);
  float* t1   = alloc((size_t)T_ * MI_);
  float* t3   = alloc((size_t)T_ * MI_);
  float* routed  = alloc((size_t)T_ * D_);
  float* sharedb = alloc((size_t)T_ * D_);
  // bf16 weights, N-major (N x K)
  unsigned short* Wt_qkv = allocU((size_t)QKVN_ * D_);
  unsigned short* Wt_ik  = allocU((size_t)DI_ * D_);
  unsigned short* Wt_iw  = allocU((size_t)HI_ * D_);
  unsigned short* Wt_qb  = allocU((size_t)QBN_ * QL_);
  unsigned short* Wt_iqb = allocU((size_t)HI_ * DI_ * QL_);
  unsigned short* Wt_ukt = allocU((size_t)H_ * KVL_ * DN_);
  unsigned short* Wt_uv  = allocU((size_t)H_ * DV_ * KVL_);
  unsigned short* Wt_o   = allocU((size_t)D_ * H_ * DV_);
  unsigned short* Wt_e1  = allocU((size_t)E_ * MI_ * D_);
  unsigned short* Wt_e3  = allocU((size_t)E_ * MI_ * D_);
  unsigned short* Wt_e2  = allocU((size_t)E_ * D_ * MI_);
  unsigned short* Wt_s1  = allocU((size_t)MI_ * D_);
  unsigned short* Wt_s3  = allocU((size_t)MI_ * D_);
  unsigned short* Wt_s2  = allocU((size_t)D_ * MI_);
  // bf16 activation shadows (A-side / B-side operands)
  unsigned short* hb   = allocU((size_t)T_ * D_);
  unsigned short* qcnb = allocU((size_t)T_ * QL_);
  unsigned short* kvb  = allocU((size_t)T_ * KVW_);        // (T x 576): B for scores
  unsigned short* kvcT = allocU((size_t)KVL_ * T_);        // (512 x T): B for ctx
  unsigned short* ikb  = allocU((size_t)T_ * DI_);         // (T x DI): B for index logits
  unsigned short* qb16 = allocU((size_t)T_ * QBN_);
  unsigned short* iq16 = allocU((size_t)T_ * HI_ * DI_);
  unsigned short* il16 = allocU((size_t)T_ * T_);
  unsigned short* qf16 = allocU((size_t)T_ * H_ * KVW_);
  unsigned short* ctx16= allocU((size_t)T_ * H_ * KVL_);
  unsigned short* ov16 = allocU((size_t)T_ * H_ * DV_);
  unsigned short* h2b  = allocU((size_t)T_ * D_);
  unsigned short* t1b  = allocU((size_t)T_ * MI_);

  const long long TD = (long long)T_ * D_;
  dim3 eb(256), eg((unsigned)((TD + 255) / 256));
  dim3 mig((unsigned)(((long long)T_ * MI_ + 255) / 256));
  auto convT = [&](const float* in, unsigned short* o, int R, int C, long long ldIn,
                   int batch = 1, long long sI = 0, long long sO = 0) {
    dim3 g((C + 31) / 32, (R + 31) / 32, batch);
    k_convT<<<g, dim3(256), 0, stream>>>(in, o, R, C, ldIn, sI, sO);
  };

  // ---- one-time weight conversion fp32 (K x N) -> bf16 (N x K) ----
  convT(W_qkv_a, Wt_qkv, D_, QKVN_, QKVN_);
  convT(W_ik,    Wt_ik,  D_, DI_,   DI_);
  convT(W_iw,    Wt_iw,  D_, HI_,   HI_);
  convT(W_qb,    Wt_qb,  QL_, QBN_, QBN_);
  convT(W_iqb,   Wt_iqb, QL_, HI_ * DI_, HI_ * DI_);
  convT(W_UKT,   Wt_ukt, DN_, KVL_, KVL_, H_, (long long)DN_ * KVL_, (long long)KVL_ * DN_);
  convT(W_UV,    Wt_uv,  KVL_, DV_, DV_,  H_, (long long)KVL_ * DV_, (long long)DV_ * KVL_);
  convT(W_o,     Wt_o,   H_ * DV_, D_, D_);
  convT(We1, Wt_e1, D_, MI_, MI_, E_, (long long)D_ * MI_, (long long)MI_ * D_);
  convT(We3, Wt_e3, D_, MI_, MI_, E_, (long long)D_ * MI_, (long long)MI_ * D_);
  convT(We2, Wt_e2, MI_, D_, D_,  E_, (long long)MI_ * D_, (long long)D_ * MI_);
  convT(Ws1, Wt_s1, D_, MI_, MI_);
  convT(Ws3, Wt_s3, D_, MI_, MI_);
  convT(Ws2, Wt_s2, MI_, D_, D_);

  // ---- pre-attention ----
  k_rmsnorm<<<T_, 256, 0, stream>>>(hidden, w_in, h_, hb, D_, D_, D_, D_);
  gemm(stream, hb, Wt_qkv, qkv, nullptr, T_, QKVN_, D_, D_, D_, QKVN_);
  k_rmsnorm<<<T_, 256, 0, stream>>>(qkv, w_qa, qcn, qcnb, QL_, QKVN_, QL_, QL_);
  k_rmsnorm<<<T_, 256, 0, stream>>>(qkv + QL_, w_kva, kv, kvb, KVL_, QKVN_, KVW_, KVW_);
  k_rope<<<dim3(T_, 1), 32, 0, stream>>>(qkv + QL_ + KVL_, kv + KVL_, kvb + KVL_, pos,
                                         QKVN_, 0, KVW_, 0, KVW_, 0, DR_, 1.f);
  gemm(stream, hb, Wt_ik, ik, nullptr, T_, DI_, D_, D_, D_, DI_);
  k_layernorm<<<T_, 256, 0, stream>>>(ik, ik_ln_w, ik_ln_b, DI_, DI_);
  k_rope<<<dim3(T_, 1), 64, 0, stream>>>(ik, ik, ikb, pos, DI_, 0, DI_, 0, DI_, 0, DI_, 1.f);
  convT(kv, kvcT, T_, KVL_, KVW_);                                 // kv_c^T bf16
  gemm(stream, hb, Wt_iw, iw, nullptr, T_, HI_, D_, D_, D_, HI_, 0.25f);  // * HI^-0.5
  gemm(stream, qcnb, Wt_qb, q, qb16, T_, QBN_, QL_, QL_, QL_, QBN_);
  k_rope<<<dim3(T_, H_), 32, 0, stream>>>(q + DN_, q + DN_, nullptr, pos,
                                          QBN_, DN_ + DR_, QBN_, DN_ + DR_, 0, 0, DR_, 1.f);
  gemm(stream, qcnb, Wt_iqb, iq, nullptr, T_, HI_ * DI_, QL_, QL_, QL_, HI_ * DI_);
  k_rope<<<dim3(T_, HI_), 64, 0, stream>>>(iq, iq, iq16, pos,
                                           HI_ * DI_, DI_, HI_ * DI_, DI_, HI_ * DI_, DI_,
                                           DI_, ISCALE_);

  // ---- index logits: il = sum_h relu(iq_h . ik^T) * iw[:,h] ----
  for (int hi = 0; hi < HI_; ++hi)
    gemm(stream, iq16 + hi * DI_, ikb, il, nullptr, T_, T_, DI_, HI_ * DI_, DI_, T_,
         1.f, /*beta=*/(hi > 0), iw + hi, HI_, /*relu=*/1);
  k_topk_mask<<<T_, 256, 0, stream>>>(il, pos, mbits);

  // ---- q_full = [q_nope . W_UKT | q_pe] (batched over heads) ----
  gemm(stream, qb16, Wt_ukt, qf, qf16, T_, KVL_, DN_, QBN_, DN_, (long long)H_ * KVW_,
       1.f, 0, nullptr, 1, 0,
       /*batch=*/H_, (long long)(DN_ + DR_), (long long)KVL_ * DN_, (long long)KVW_);
  k_copy_qpe<<<dim3(T_, H_), 64, 0, stream>>>(q, qf, qf16);

  // ---- attention per head: scores -> masked softmax -> ctx ----
  for (int h = 0; h < H_; ++h) {
    gemm(stream, qf16 + h * KVW_, kvb, il, nullptr, T_, T_, KVW_,
         (long long)H_ * KVW_, KVW_, T_, SCALE_);
    k_softmax_mask<<<T_, 256, 0, stream>>>(il, il16, mbits);
    gemm(stream, il16, kvcT, ctx + h * KVL_, ctx16 + h * KVL_,
         T_, KVL_, T_, T_, T_, (long long)H_ * KVL_);
  }

  // ---- output projection + residual ----
  gemm(stream, ctx16, Wt_uv, ov, ov16, T_, DV_, KVL_,
       (long long)H_ * KVL_, KVL_, (long long)H_ * DV_,
       1.f, 0, nullptr, 1, 0,
       /*batch=*/H_, (long long)KVL_, (long long)DV_ * KVL_, (long long)DV_);
  gemm(stream, ov16, Wt_o, oproj, nullptr, T_, D_, H_ * DV_, H_ * DV_, H_ * DV_, D_);
  k_add<<<eg, eb, 0, stream>>>(oproj, hidden, resid, TD);

  // ---- MoE ----
  k_rmsnorm<<<T_, 256, 0, stream>>>(resid, w_post, h2, h2b, D_, D_, D_, D_);
  k_gate<<<T_, 256, 0, stream>>>(h2, W_gate, we);
  for (int e = 0; e < E_; ++e) {
    const unsigned short* e1 = Wt_e1 + (long long)e * MI_ * D_;
    const unsigned short* e3 = Wt_e3 + (long long)e * MI_ * D_;
    const unsigned short* e2 = Wt_e2 + (long long)e * D_ * MI_;
    gemm(stream, h2b, e1, t1, nullptr, T_, MI_, D_, D_, D_, MI_);
    gemm(stream, h2b, e3, t3, nullptr, T_, MI_, D_, D_, D_, MI_);
    k_silu_mul<<<mig, eb, 0, stream>>>(t1, t3, t1b, (long long)T_ * MI_);
    gemm(stream, t1b, e2, routed, nullptr, T_, D_, MI_, MI_, MI_, D_,
         1.f, /*beta=*/(e > 0), we + e, E_, 0);
  }
  gemm(stream, h2b, Wt_s1, t1, nullptr, T_, MI_, D_, D_, D_, MI_);
  gemm(stream, h2b, Wt_s3, t3, nullptr, T_, MI_, D_, D_, D_, MI_);
  k_silu_mul<<<mig, eb, 0, stream>>>(t1, t3, t1b, (long long)T_ * MI_);
  gemm(stream, t1b, Wt_s2, sharedb, nullptr, T_, D_, MI_, MI_, MI_, D_);
  k_combine<<<eg, eb, 0, stream>>>(routed, sharedb, out, TD);
}
